// MPGNOCust_87986700026102
// MI455X (gfx1250) — matrix operand order repaired and checked by reading, not compiled, not run on hardware
//
#include <hip/hip_runtime.h>

#define NNODES 16384
#define NEDGES 32768
#define LDIM   32
#define KDIM   64
#define NLAYER 14          // 7 blocks * 2 layers
#define ETILES (NEDGES/16) // 2048 edge tiles of 16 edges

typedef __attribute__((ext_vector_type(16))) _Float16 v16h;
typedef __attribute__((ext_vector_type(8)))  float    v8f;

__device__ __forceinline__ float geluf(float x) {
    return 0.5f * x * (1.0f + erff(x * 0.7071067811865475f));
}

// ---------------------------------------------------------------------------
// Projector: x = gelu( gelu([nodes,grid] @ w1 + b1) @ w2 + b2 )   (12->16->32)
// ---------------------------------------------------------------------------
__global__ void proj_kernel(const float* __restrict__ nodes, const float* __restrict__ grid,
                            const float* __restrict__ w1, const float* __restrict__ b1,
                            const float* __restrict__ w2, const float* __restrict__ b2,
                            float* __restrict__ xout) {
    int n = blockIdx.x * blockDim.x + threadIdx.x;
    if (n >= NNODES) return;
    float in[12];
#pragma unroll
    for (int t = 0; t < 10; t++) in[t] = nodes[n * 10 + t];
    in[10] = grid[n * 2 + 0];
    in[11] = grid[n * 2 + 1];
    float h[16];
#pragma unroll
    for (int j = 0; j < 16; j++) {
        float s = b1[j];
#pragma unroll
        for (int t = 0; t < 12; t++) s += in[t] * w1[t * 16 + j];
        h[j] = geluf(s);
    }
#pragma unroll
    for (int o = 0; o < 32; o++) {
        float s = b2[o];
#pragma unroll
        for (int j = 0; j < 16; j++) s += h[j] * w2[j * 32 + o];
        xout[n * 32 + o] = geluf(s);
    }
}

// ---------------------------------------------------------------------------
// Degree: count dst occurrences then invert (1/deg, 0 if deg==0)
// ---------------------------------------------------------------------------
__global__ void deg_count_kernel(const int* __restrict__ eidx, float* __restrict__ deg) {
    int e = blockIdx.x * blockDim.x + threadIdx.x;
    if (e < NEDGES) atomicAdd(&deg[eidx[NEDGES + e]], 1.0f);
}
__global__ void deg_finish_kernel(float* __restrict__ deg, int n) {
    int i = blockIdx.x * blockDim.x + threadIdx.x;
    if (i < n) { float d = deg[i]; deg[i] = (d > 0.0f) ? (1.0f / d) : 0.0f; }
}

// ---------------------------------------------------------------------------
// Repack kw2 (f32 [14][64][1024]) into f16 WMMA B-fragment order:
//   packed[layer][nt(64)][ks(2)][lane(32)][t(16)]
//   K = ks*32 + (lane<16 ? t : 16+t), Ncol = nt*16 + (lane&15)
// ---------------------------------------------------------------------------
__global__ void repack_kw2_kernel(const float* __restrict__ kw2, _Float16* __restrict__ packed) {
    int idx = blockIdx.x * blockDim.x + threadIdx.x;
    if (idx >= NLAYER * 65536) return;
    int t     = idx & 15;
    int lane  = (idx >> 4) & 31;
    int ks    = (idx >> 9) & 1;
    int nt    = (idx >> 10) & 63;
    int layer = idx >> 16;
    int K  = ks * 32 + t + ((lane >> 4) << 4);
    int Nc = nt * 16 + (lane & 15);
    packed[idx] = (_Float16)kw2[((size_t)(layer * 64 + K)) * 1024 + Nc];
}

// ---------------------------------------------------------------------------
// Fused edge kernel (512 threads = 16 waves): per 16-edge tile,
//   h = gelu(eattr @ kw1 + kb1)   [16 x 64], stored in A-fragment order in LDS
//   k = h @ kw2 + kb2             via v_wmma_f32_16x16x32_f16 (2 k-steps)
//   msg[e,o] = sum_i k[e, i*32+o] * x[src(e),i]
//   atomicAdd agg[dst(e)*32+o] += msg[e,o]
// Wave w owns N-tiles {w, w+16, w+32, w+48}; tile parity fixed per wave, so
// each lane's output column o is constant and only i = (w>>1)+8j varies.
// ---------------------------------------------------------------------------
__global__ void __launch_bounds__(512, 1)
gno_edge_kernel(const float* __restrict__ eattr, const int* __restrict__ eidx,
                const float* __restrict__ x,
                const float* __restrict__ kw1, const float* __restrict__ kb1,
                const _Float16* __restrict__ kw2p, const float* __restrict__ kb2,
                float* __restrict__ agg) {
    __shared__ __align__(32) _Float16 hfrag[2 * 32 * 16]; // [ks][lane][t], A-fragment order
    __shared__ __align__(16) float    xsT[32 * 16];       // [i][m] transposed gather
    __shared__ float                  msg_lds[16 * 32];   // [m][o]

    const int tid  = threadIdx.x;
    const int lane = tid & 31;
    const int w    = tid >> 5;      // 0..15

    // Per-wave B fragments (4 tiles x 2 k-steps = 64 VGPRs), loaded once.
    const v16h* Bp = (const v16h*)kw2p;
    v16h  Breg[4][2];
    float biasj[4];
#pragma unroll
    for (int j = 0; j < 4; j++) {
        const int nt = w + 16 * j;
#pragma unroll
        for (int ks = 0; ks < 2; ks++)
            Breg[j][ks] = Bp[(nt * 2 + ks) * 32 + lane];
        biasj[j] = kb2[nt * 16 + (lane & 15)];
    }

    const int mbase = (lane < 16) ? 0 : 8;
    const int ocol  = (w & 1) * 16 + (lane & 15);

    for (int et = blockIdx.x; et < ETILES; et += gridDim.x) {
        const int base = et * 16;

        msg_lds[tid & 511] = 0.0f;

        // h = gelu(eattr @ kw1 + kb1); store directly in A-fragment order.
        {
            const int m  = tid >> 5;          // edge row 0..15
            const int j0 = (lane) * 2;        // 2 columns of KD=64 per thread
            const int e  = base + m;
            const float a0 = eattr[e * 5 + 0], a1 = eattr[e * 5 + 1], a2 = eattr[e * 5 + 2];
            const float a3 = eattr[e * 5 + 3], a4 = eattr[e * 5 + 4];
#pragma unroll
            for (int jj = 0; jj < 2; jj++) {
                const int j = j0 + jj;
                float s = kb1[j] + a0 * kw1[0 * 64 + j] + a1 * kw1[1 * 64 + j] +
                          a2 * kw1[2 * 64 + j] + a3 * kw1[3 * 64 + j] + a4 * kw1[4 * 64 + j];
                const int ks    = j >> 5;
                const int K     = j & 31;
                const int laneH = m + ((K >> 4) << 4);
                const int t     = K & 15;
                hfrag[(ks * 32 + laneH) * 16 + t] = (_Float16)geluf(s);
            }
            // transposed gather of x[src]: xsT[i][m]
            const int src = eidx[e];
            xsT[lane * 16 + m] = x[(size_t)src * 32 + lane];
        }
        __syncthreads();

        // contiguous A-fragment loads (2x ds_load_b128 each)
        v16h Af0 = *(const v16h*)&hfrag[(0 * 32 + lane) * 16];
        v16h Af1 = *(const v16h*)&hfrag[(1 * 32 + lane) * 16];

        float partial[8];
#pragma unroll
        for (int v = 0; v < 8; v++) partial[v] = 0.0f;

#pragma unroll
        for (int j = 0; j < 4; j++) {
            v8f c;
#pragma unroll
            for (int v = 0; v < 8; v++) c[v] = biasj[j];
            c = __builtin_amdgcn_wmma_f32_16x16x32_f16(false, Af0, false, Breg[j][0],
                                                       (short)0, c, false, false);
            c = __builtin_amdgcn_wmma_f32_16x16x32_f16(false, Af1, false, Breg[j][1],
                                                       (short)0, c, false, false);
            const int i = (w >> 1) + 8 * j;   // latent input index for this tile
            const float4 xa = *(const float4*)&xsT[i * 16 + mbase];
            const float4 xb = *(const float4*)&xsT[i * 16 + mbase + 4];
            partial[0] += c[0] * xa.x; partial[1] += c[1] * xa.y;
            partial[2] += c[2] * xa.z; partial[3] += c[3] * xa.w;
            partial[4] += c[4] * xb.x; partial[5] += c[5] * xb.y;
            partial[6] += c[6] * xb.z; partial[7] += c[7] * xb.w;
        }
        // reduce same-parity waves into msg_lds
#pragma unroll
        for (int v = 0; v < 8; v++)
            atomicAdd(&msg_lds[(mbase + v) * 32 + ocol], partial[v]);
        __syncthreads();

        // scatter msg to agg[dst] (one value per thread)
        {
            const int m = tid >> 5, o = tid & 31;
            const int dst = eidx[NEDGES + base + m];
            atomicAdd(&agg[(size_t)dst * 32 + o], msg_lds[tid]);
        }
        __syncthreads();
    }
}

// ---------------------------------------------------------------------------
// Node update: x_new = gelu(x @ lw + lb + agg * inv_deg)
// ---------------------------------------------------------------------------
__global__ void gno_node_kernel(const float* __restrict__ xin, const float* __restrict__ agg,
                                const float* __restrict__ invdeg,
                                const float* __restrict__ lw, const float* __restrict__ lb,
                                float* __restrict__ xout) {
    int idx = blockIdx.x * blockDim.x + threadIdx.x;
    if (idx >= NNODES * 32) return;
    int n = idx >> 5, o = idx & 31;
    float s = lb[o] + agg[idx] * invdeg[n];
    const float* xr = xin + (size_t)n * 32;
#pragma unroll
    for (int i = 0; i < 32; i++) s += xr[i] * lw[i * 32 + o];
    xout[idx] = geluf(s);
}

__global__ void add_kernel(const float* __restrict__ a, const float* __restrict__ b,
                           float* __restrict__ c, int n) {
    int i = blockIdx.x * blockDim.x + threadIdx.x;
    if (i < n) c[i] = a[i] + b[i];
}

// ---------------------------------------------------------------------------
// Decoder: out = gelu((n21+n11) @ dw1 + db1) @ dw2 + db2   (32->16->1)
// ---------------------------------------------------------------------------
__global__ void dec_kernel(const float* __restrict__ a, const float* __restrict__ b,
                           const float* __restrict__ w1, const float* __restrict__ b1,
                           const float* __restrict__ w2, const float* __restrict__ b2,
                           float* __restrict__ out) {
    int n = blockIdx.x * blockDim.x + threadIdx.x;
    if (n >= NNODES) return;
    float in[32];
#pragma unroll
    for (int i = 0; i < 32; i++) in[i] = a[n * 32 + i] + b[n * 32 + i];
    float s_out = b2[0];
#pragma unroll
    for (int j = 0; j < 16; j++) {
        float s = b1[j];
#pragma unroll
        for (int i = 0; i < 32; i++) s += in[i] * w1[i * 16 + j];
        s_out += geluf(s) * w2[j];
    }
    out[n] = s_out;
}

// ---------------------------------------------------------------------------
extern "C" void kernel_launch(void* const* d_in, const int* in_sizes, int n_in,
                              void* d_out, int out_size, void* d_ws, size_t ws_size,
                              hipStream_t stream) {
    (void)in_sizes; (void)n_in; (void)out_size; (void)ws_size;
    const float* nodes = (const float*)d_in[0];
    const float* gridp = (const float*)d_in[1];
    const int*   eidx[7];
    const float* eattr[7];
    for (int i = 0; i < 7; i++) { eidx[i] = (const int*)d_in[2 + i]; eattr[i] = (const float*)d_in[9 + i]; }
    const float* pw1 = (const float*)d_in[18]; const float* pb1 = (const float*)d_in[19];
    const float* pw2 = (const float*)d_in[20]; const float* pb2 = (const float*)d_in[21];
    const float* dw1 = (const float*)d_in[22]; const float* db1 = (const float*)d_in[23];
    const float* dw2 = (const float*)d_in[24]; const float* db2 = (const float*)d_in[25];
    const float* kw1 = (const float*)d_in[26]; const float* kb1 = (const float*)d_in[27];
    const float* kw2 = (const float*)d_in[28]; const float* kb2 = (const float*)d_in[29];
    const float* lw  = (const float*)d_in[30]; const float* lb  = (const float*)d_in[31];

    char* ws = (char*)d_ws;
    size_t off = 0;
    auto alloc = [&](size_t bytes) -> char* {
        char* p = ws + off;
        off += (bytes + 255) & ~(size_t)255;
        return p;
    };
    const size_t NB = (size_t)NNODES * 32 * sizeof(float);
    float* x0   = (float*)alloc(NB);
    float* n11  = (float*)alloc(NB);
    float* n12  = (float*)alloc(NB);
    float* n22  = (float*)alloc(NB);
    float* n23  = (float*)alloc(NB);
    float* n33  = (float*)alloc(NB);
    float* n32  = (float*)alloc(NB);
    float* nsum = (float*)alloc(NB);
    float* n21  = (float*)alloc(NB);
    float* xtmp = (float*)alloc(NB);
    float* agg  = (float*)alloc(NB);
    float* invdeg = (float*)alloc((size_t)7 * NNODES * sizeof(float));
    _Float16* kw2p = (_Float16*)alloc((size_t)NLAYER * 65536 * sizeof(_Float16));

    // degrees (7 edge sets)
    hipMemsetAsync(invdeg, 0, (size_t)7 * NNODES * sizeof(float), stream);
    for (int ei = 0; ei < 7; ei++)
        deg_count_kernel<<<NEDGES / 256, 256, 0, stream>>>(eidx[ei], invdeg + (size_t)ei * NNODES);
    deg_finish_kernel<<<(7 * NNODES) / 256, 256, 0, stream>>>(invdeg, 7 * NNODES);

    // weight repack + projector
    repack_kw2_kernel<<<(NLAYER * 65536) / 256, 256, 0, stream>>>(kw2, kw2p);
    proj_kernel<<<NNODES / 256, 256, 0, stream>>>(nodes, gridp, pw1, pb1, pw2, pb2, x0);

    auto run_block = [&](int pi, int ei, const float* xin, float* xout) {
        for (int d = 0; d < 2; d++) {
            const int li = pi * 2 + d;
            const float* xi = (d == 0) ? xin : xtmp;
            float* xo = (d == 0) ? xtmp : xout;
            hipMemsetAsync(agg, 0, NB, stream);
            gno_edge_kernel<<<512, 512, 0, stream>>>(
                eattr[ei], eidx[ei], xi,
                kw1 + (size_t)li * 320, kb1 + (size_t)li * 64,
                kw2p + (size_t)li * 65536, kb2 + (size_t)li * 1024, agg);
            gno_node_kernel<<<(NNODES * 32) / 256, 256, 0, stream>>>(
                xi, agg, invdeg + (size_t)ei * NNODES,
                lw + (size_t)li * 1024, lb + (size_t)li * 32, xo);
        }
    };

    run_block(0, 0, x0,  n11);   // k11
    run_block(3, 1, x0,  n12);   // k12
    run_block(1, 2, n12, n22);   // k22
    run_block(4, 3, n12, n23);   // k23
    run_block(2, 4, n23, n33);   // k33
    run_block(5, 5, n33, n32);   // k32
    add_kernel<<<(NNODES * 32) / 256, 256, 0, stream>>>(n32, n22, nsum, NNODES * 32);
    run_block(6, 6, nsum, n21);  // k21

    dec_kernel<<<NNODES / 256, 256, 0, stream>>>(n21, n11, dw1, db1, dw2, db2, (float*)d_out);
}
